// HeterogeneousNormalizedAttention_26431228739642
// MI455X (gfx1250) — matrix operand order repaired and checked by reading, not compiled, not run on hardware
//
#include <hip/hip_runtime.h>
#include <hip/hip_bf16.h>

typedef __attribute__((ext_vector_type(16))) _Float16 v16h;
typedef __attribute__((ext_vector_type(8)))  _Float16 v8h;
typedef __attribute__((ext_vector_type(4)))  _Float16 v4h;
typedef __attribute__((ext_vector_type(8)))  float    v8f;

#define D_MODEL 512
#define NHEAD   8
#define HD      64
#define NI      3
#define BATCH   4
#define NSEQ    8192
#define NIN     4096
#define MROWS_X (BATCH * NSEQ)   // 32768
#define MROWS_E (BATCH * NIN)    // 16384

// ---------------------------------------------------------------------------
// helpers
// ---------------------------------------------------------------------------
static __device__ __forceinline__ v16h ld16pair(const _Float16* lo, const _Float16* hi) {
  v8h a = *(const v8h*)lo;
  v8h b = *(const v8h*)hi;
  return __builtin_shufflevector(a, b, 0,1,2,3,4,5,6,7,8,9,10,11,12,13,14,15);
}

static __device__ __forceinline__ v8f wmma_f16(v16h a, v16h b, v8f c) {
  // (neg_a, A, neg_b, B, c_mod, C, reuse_a, reuse_b)
  return __builtin_amdgcn_wmma_f32_16x16x32_f16(false, a, false, b, (short)0, c, false, false);
}

static __device__ __forceinline__ v8f vzero8() {
  v8f z = {0.f,0.f,0.f,0.f,0.f,0.f,0.f,0.f};
  return z;
}

// ---------------------------------------------------------------------------
// fp32 -> fp16 convert (vectorized x4)
// ---------------------------------------------------------------------------
__global__ __launch_bounds__(256) void cvt_fp32_to_fp16(const float* __restrict__ in,
                                                        _Float16* __restrict__ out, int n4) {
  int i = blockIdx.x * blockDim.x + threadIdx.x;
  if (i < n4) {
    float4 v = ((const float4*)in)[i];
    v4h h;
    h[0] = (_Float16)v.x; h[1] = (_Float16)v.y;
    h[2] = (_Float16)v.z; h[3] = (_Float16)v.w;
    ((v4h*)out)[i] = h;
  }
}

// ---------------------------------------------------------------------------
// Generic WMMA GEMM: C[M,512] = A[M,512] @ W^T (+bias), W row-major [512][512]
// Each wave computes a 32-row x 64-col strip (one head-width). The K loop is
// FULLY UNROLLED with one-step-ahead prefetch: iteration k's operand loads are
// issued before iteration k-1's wmmas, so the per-wmma s_wait_loadcnt is
// partial and load latency overlaps compute; full unroll turns the buffer
// rotation into SSA renames (no v_mov copies).
// mode 0: +bias, store f16
// mode 1: +bias, per-row softmax over 64 cols, store f16      (q projection)
// mode 2: mode 1 then multiply row by mask[row]                (k projection)
// mode 3: +bias, store fp32                                    (output proj)
// ---------------------------------------------------------------------------
__global__ __launch_bounds__(256) void gemm_kernel(const _Float16* __restrict__ A,
                                                   const _Float16* __restrict__ W,
                                                   const float* __restrict__ bias,
                                                   _Float16* __restrict__ outH,
                                                   float* __restrict__ outF,
                                                   const unsigned char* __restrict__ mask,
                                                   int mode) {
  const int lane = threadIdx.x & 31;
  const int wave = threadIdx.x >> 5;
  const int ln   = lane & 15;
  const int sel  = lane >> 4;
  const int m0 = (blockIdx.x * 8 + wave) * 32;
  const int n0 = blockIdx.y * 64;

  v8f acc[2][4];
#pragma unroll
  for (int g = 0; g < 2; ++g)
#pragma unroll
    for (int t = 0; t < 4; ++t) acc[g][t] = vzero8();

  const _Float16* Arow0 = A + (size_t)(m0 + ln) * D_MODEL + sel * 8;
  const _Float16* Arow1 = A + (size_t)(m0 + 16 + ln) * D_MODEL + sel * 8;
  const _Float16* Wbase[4];
#pragma unroll
  for (int t = 0; t < 4; ++t)
    Wbase[t] = W + (size_t)(n0 + t * 16 + ln) * D_MODEL + sel * 8;

  // prologue loads (k0 = 0)
  v16h a0 = ld16pair(Arow0, Arow0 + 16);
  v16h a1 = ld16pair(Arow1, Arow1 + 16);
  v16h bb[4];
#pragma unroll
  for (int t = 0; t < 4; ++t) bb[t] = ld16pair(Wbase[t], Wbase[t] + 16);

#pragma unroll
  for (int k0 = 0; k0 < D_MODEL; k0 += 32) {
    v16h na0, na1, nb[4];
    const int kn = k0 + 32;
    if (kn < D_MODEL) {               // statically resolved under full unroll
      na0 = ld16pair(Arow0 + kn, Arow0 + kn + 16);
      na1 = ld16pair(Arow1 + kn, Arow1 + kn + 16);
#pragma unroll
      for (int t = 0; t < 4; ++t) nb[t] = ld16pair(Wbase[t] + kn, Wbase[t] + kn + 16);
    }
#pragma unroll
    for (int t = 0; t < 4; ++t) acc[0][t] = wmma_f16(a0, bb[t], acc[0][t]);
#pragma unroll
    for (int t = 0; t < 4; ++t) acc[1][t] = wmma_f16(a1, bb[t], acc[1][t]);
    if (kn < D_MODEL) {
      a0 = na0; a1 = na1;
#pragma unroll
      for (int t = 0; t < 4; ++t) bb[t] = nb[t];
    }
  }

  // bias
  float bvals[4];
#pragma unroll
  for (int t = 0; t < 4; ++t) bvals[t] = bias[n0 + t * 16 + ln];
#pragma unroll
  for (int g = 0; g < 2; ++g)
#pragma unroll
    for (int t = 0; t < 4; ++t)
#pragma unroll
      for (int r = 0; r < 8; ++r) acc[g][t][r] += bvals[t];

  if (mode == 1 || mode == 2) {
    // per-row softmax over the 64 columns of this strip.
    // Row m0+g*16+r+8*sel lives in the 16 lanes of this half-wave, 4 cols/lane.
#pragma unroll
    for (int g = 0; g < 2; ++g)
#pragma unroll
      for (int r = 0; r < 8; ++r) {
        float mx = acc[g][0][r];
#pragma unroll
        for (int t = 1; t < 4; ++t) mx = fmaxf(mx, acc[g][t][r]);
        for (int off = 1; off < 16; off <<= 1) mx = fmaxf(mx, __shfl_xor(mx, off, 32));
        float e[4]; float s = 0.f;
#pragma unroll
        for (int t = 0; t < 4; ++t) { e[t] = __expf(acc[g][t][r] - mx); s += e[t]; }
        for (int off = 1; off < 16; off <<= 1) s += __shfl_xor(s, off, 32);
        float inv = 1.f / s;
#pragma unroll
        for (int t = 0; t < 4; ++t) acc[g][t][r] = e[t] * inv;
      }
  }
  if (mode == 2) {
#pragma unroll
    for (int g = 0; g < 2; ++g)
#pragma unroll
      for (int r = 0; r < 8; ++r) {
        int row = m0 + g * 16 + r + sel * 8;
        float sc = mask[row] ? 1.f : 0.f;
#pragma unroll
        for (int t = 0; t < 4; ++t) acc[g][t][r] *= sc;
      }
  }

#pragma unroll
  for (int g = 0; g < 2; ++g)
#pragma unroll
    for (int r = 0; r < 8; ++r) {
      int row = m0 + g * 16 + r + sel * 8;
#pragma unroll
      for (int t = 0; t < 4; ++t) {
        int col = n0 + t * 16 + ln;
        if (mode == 3) outF[(size_t)row * D_MODEL + col] = acc[g][t][r];
        else           outH[(size_t)row * D_MODEL + col] = (_Float16)acc[g][t][r];
      }
    }
}

// ---------------------------------------------------------------------------
// kv kernel: per (i,b,h) compute kv[d][e] = sum_s k[s,d]*v[s,e]  (K = 4096)
// Writes kvT[e][d] (transposed) so q@kv B-operand is contiguous later.
// LDS transpose staging; 8 waves x 2 WMMA tiles cover the 64x64 output.
// ---------------------------------------------------------------------------
#define KV_SCHUNK 128
#define KV_PITCH  136   // padded, multiple of 8 halves (16B)

__global__ __launch_bounds__(256) void kv_kernel(const _Float16* __restrict__ kH,
                                                 const _Float16* __restrict__ vH,
                                                 _Float16* __restrict__ kvT) {
  __shared__ _Float16 kT[64 * KV_PITCH];
  __shared__ _Float16 vT[64 * KV_PITCH];

  const int blk = blockIdx.x;                 // 0..95
  const int i = blk / (BATCH * NHEAD);
  const int rem = blk % (BATCH * NHEAD);
  const int b = rem / NHEAD, h = rem % NHEAD;
  const size_t gbase = (size_t)i * MROWS_E * D_MODEL + (size_t)b * NIN * D_MODEL + h * HD;

  const int tid = threadIdx.x;
  const int lane = tid & 31, wave = tid >> 5;
  const int ln = lane & 15, sel = lane >> 4;
  const int dm = (wave >> 1) * 16;            // row-tile (d)
  const int e0 = (wave & 1) * 32;             // first col-tile (e)

  v8f acc[2];
  acc[0] = vzero8(); acc[1] = vzero8();

  for (int s0 = 0; s0 < NIN; s0 += KV_SCHUNK) {
    // stage with transpose: LDS[d][s]
    for (int u = tid; u < KV_SCHUNK * 64; u += 256) {
      int s = u >> 6, d = u & 63;
      kT[d * KV_PITCH + s] = kH[gbase + (size_t)(s0 + s) * D_MODEL + d];
      vT[d * KV_PITCH + s] = vH[gbase + (size_t)(s0 + s) * D_MODEL + d];
    }
    __syncthreads();
#pragma unroll
    for (int sk = 0; sk < KV_SCHUNK; sk += 32) {
      const _Float16* ap = kT + (dm + ln) * KV_PITCH + sk + sel * 8;
      v16h a = ld16pair(ap, ap + 16);
#pragma unroll
      for (int t = 0; t < 2; ++t) {
        const _Float16* bp = vT + (e0 + t * 16 + ln) * KV_PITCH + sk + sel * 8;
        v16h bm = ld16pair(bp, bp + 16);
        acc[t] = wmma_f16(a, bm, acc[t]);
      }
    }
    __syncthreads();
  }

  _Float16* outb = kvT + (size_t)((i * BATCH + b) * NHEAD + h) * HD * HD;
#pragma unroll
  for (int t = 0; t < 2; ++t)
#pragma unroll
    for (int r = 0; r < 8; ++r) {
      int d = dm + r + sel * 8;
      int e = e0 + t * 16 + ln;
      outb[(size_t)e * HD + d] = (_Float16)acc[t][r];   // transposed store
    }
}

// ---------------------------------------------------------------------------
// k_sum: per (i,b,h,d) sum over s of masked k
// ---------------------------------------------------------------------------
__global__ __launch_bounds__(256) void ksum_kernel(const _Float16* __restrict__ kH,
                                                   float* __restrict__ ksum) {
  const int blk = blockIdx.x;                 // 96 blocks
  const int i = blk / (BATCH * NHEAD);
  const int rem = blk % (BATCH * NHEAD);
  const int b = rem / NHEAD, h = rem % NHEAD;
  const int t = threadIdx.x;
  const int d = t & 63, part = t >> 6;
  const _Float16* base = kH + (size_t)i * MROWS_E * D_MODEL +
                         (size_t)b * NIN * D_MODEL + h * HD + d;
  float s = 0.f;
  for (int sdx = part; sdx < NIN; sdx += 4) s += (float)base[(size_t)sdx * D_MODEL];
  __shared__ float red[256];
  red[t] = s;
  __syncthreads();
  if (t < 64)
    ksum[((size_t)(i * BATCH + b) * NHEAD + h) * HD + d] =
        red[t] + red[t + 64] + red[t + 128] + red[t + 192];
}

// ---------------------------------------------------------------------------
// d_inv[i, row, h] = 1 / dot(q[row, h*64: ], ksum[i,b,h,:])
// one thread per (row, h)
// ---------------------------------------------------------------------------
__global__ __launch_bounds__(256) void dinv_kernel(const _Float16* __restrict__ qH,
                                                   const float* __restrict__ ksum,
                                                   float* __restrict__ dinv) {
  const int idx = blockIdx.x * blockDim.x + threadIdx.x;  // row*8 + h
  const int h = idx & 7;
  const int row = idx >> 3;
  const int b = row >> 13;                                // / 8192
  const _Float16* qp = qH + (size_t)row * D_MODEL + h * HD;
  float dot[NI] = {0.f, 0.f, 0.f};
  for (int j = 0; j < HD; j += 8) {
    v8h qv = *(const v8h*)(qp + j);
#pragma unroll
    for (int i = 0; i < NI; ++i) {
      const float* ks = ksum + ((size_t)(i * BATCH + b) * NHEAD + h) * HD + j;
#pragma unroll
      for (int e = 0; e < 8; ++e) dot[i] += (float)qv[e] * ks[e];
    }
  }
#pragma unroll
  for (int i = 0; i < NI; ++i)
    dinv[((size_t)i * MROWS_X + row) * NHEAD + h] = 1.0f / dot[i];
}

// ---------------------------------------------------------------------------
// combine: out = (q + sum_i (q @ kv_i) * d_inv_i) / 3, store f16 [32768,512]
// wave per (16 rows, one head)
// ---------------------------------------------------------------------------
__global__ __launch_bounds__(256) void combine_kernel(const _Float16* __restrict__ qH,
                                                      const _Float16* __restrict__ kvT,
                                                      const float* __restrict__ dinv,
                                                      _Float16* __restrict__ outH) {
  const int lane = threadIdx.x & 31, wave = threadIdx.x >> 5;
  const int ln = lane & 15, sel = lane >> 4;
  const int strip = blockIdx.x * 8 + wave;
  const int h = strip & 7, rt = strip >> 3;
  const int m0 = rt * 16;
  const int b = m0 >> 13;

  const _Float16* qrow = qH + (size_t)(m0 + ln) * D_MODEL + h * HD;
  v16h a0 = ld16pair(qrow + sel * 8,      qrow + sel * 8 + 16);
  v16h a1 = ld16pair(qrow + 32 + sel * 8, qrow + 32 + sel * 8 + 16);

  v8f out[4];
#pragma unroll
  for (int t = 0; t < 4; ++t)
#pragma unroll
    for (int r = 0; r < 8; ++r) {
      int row = m0 + r + sel * 8;
      out[t][r] = (float)qH[(size_t)row * D_MODEL + h * HD + t * 16 + ln];
    }

  for (int i = 0; i < NI; ++i) {
    const _Float16* kb = kvT + (size_t)((i * BATCH + b) * NHEAD + h) * HD * HD;
    v8f acc[4];
#pragma unroll
    for (int t = 0; t < 4; ++t) acc[t] = vzero8();
#pragma unroll
    for (int t = 0; t < 4; ++t) {
      const _Float16* bp0 = kb + (size_t)(t * 16 + ln) * HD + sel * 8;
      v16h b0 = ld16pair(bp0, bp0 + 16);
      acc[t] = wmma_f16(a0, b0, acc[t]);
      const _Float16* bp1 = kb + (size_t)(t * 16 + ln) * HD + 32 + sel * 8;
      v16h b1 = ld16pair(bp1, bp1 + 16);
      acc[t] = wmma_f16(a1, b1, acc[t]);
    }
#pragma unroll
    for (int r = 0; r < 8; ++r) {
      int row = m0 + r + sel * 8;
      float dv = dinv[((size_t)i * MROWS_X + row) * NHEAD + h];
#pragma unroll
      for (int t = 0; t < 4; ++t) out[t][r] += acc[t][r] * dv;
    }
  }

  const float inv3 = 1.0f / 3.0f;
#pragma unroll
  for (int r = 0; r < 8; ++r) {
    int row = m0 + r + sel * 8;
#pragma unroll
    for (int t = 0; t < 4; ++t)
      outH[(size_t)row * D_MODEL + h * HD + t * 16 + ln] = (_Float16)(out[t][r] * inv3);
  }
}

// ---------------------------------------------------------------------------
// launch
// ---------------------------------------------------------------------------
extern "C" void kernel_launch(void* const* d_in, const int* in_sizes, int n_in,
                              void* d_out, int out_size, void* d_ws, size_t ws_size,
                              hipStream_t stream) {
  (void)in_sizes; (void)n_in; (void)out_size; (void)ws_size;
  const float* x    = (const float*)d_in[0];
  const float* emb0 = (const float*)d_in[1];
  const float* emb1 = (const float*)d_in[2];
  const float* emb2 = (const float*)d_in[3];
  const float* Wq   = (const float*)d_in[4];
  const float* bq   = (const float*)d_in[5];
  const float* Wk   = (const float*)d_in[6];
  const float* bk   = (const float*)d_in[7];
  const float* Wv   = (const float*)d_in[8];
  const float* bv   = (const float*)d_in[9];
  const float* Wo   = (const float*)d_in[10];
  const float* bo   = (const float*)d_in[11];
  const unsigned char* masks[NI] = {(const unsigned char*)d_in[12],
                                    (const unsigned char*)d_in[13],
                                    (const unsigned char*)d_in[14]};
  const float* embs[NI] = {emb0, emb1, emb2};

  char* ws = (char*)d_ws;
  size_t off = 0;
  auto alloc = [&](size_t bytes) -> char* {
    char* p = ws + off;
    off += (bytes + 255) & ~(size_t)255;
    return p;
  };

  _Float16* xh   = (_Float16*)alloc((size_t)MROWS_X * D_MODEL * 2);
  _Float16* eh   = (_Float16*)alloc((size_t)NI * MROWS_E * D_MODEL * 2);
  _Float16* qh   = (_Float16*)alloc((size_t)MROWS_X * D_MODEL * 2);
  _Float16* kh   = (_Float16*)alloc((size_t)NI * MROWS_E * D_MODEL * 2);
  _Float16* vh   = (_Float16*)alloc((size_t)NI * MROWS_E * D_MODEL * 2);
  _Float16* oh   = (_Float16*)alloc((size_t)MROWS_X * D_MODEL * 2);
  _Float16* Wqh  = (_Float16*)alloc((size_t)D_MODEL * D_MODEL * 2);
  _Float16* Wkh  = (_Float16*)alloc((size_t)NI * D_MODEL * D_MODEL * 2);
  _Float16* Wvh  = (_Float16*)alloc((size_t)NI * D_MODEL * D_MODEL * 2);
  _Float16* Woh  = (_Float16*)alloc((size_t)D_MODEL * D_MODEL * 2);
  _Float16* kvT  = (_Float16*)alloc((size_t)NI * BATCH * NHEAD * HD * HD * 2);
  float*    ksum = (float*)alloc((size_t)NI * BATCH * NHEAD * HD * 4);
  float*    dinv = (float*)alloc((size_t)NI * MROWS_X * NHEAD * 4);

  auto cvt = [&](const float* src, _Float16* dst, size_t n) {
    int n4 = (int)(n / 4);
    cvt_fp32_to_fp16<<<(n4 + 255) / 256, 256, 0, stream>>>(src, dst, n4);
  };

  cvt(x, xh, (size_t)MROWS_X * D_MODEL);
  for (int i = 0; i < NI; ++i)
    cvt(embs[i], eh + (size_t)i * MROWS_E * D_MODEL, (size_t)MROWS_E * D_MODEL);
  cvt(Wq, Wqh, (size_t)D_MODEL * D_MODEL);
  cvt(Wk, Wkh, (size_t)NI * D_MODEL * D_MODEL);
  cvt(Wv, Wvh, (size_t)NI * D_MODEL * D_MODEL);
  cvt(Wo, Woh, (size_t)D_MODEL * D_MODEL);

  // q = softmax(x @ Wq^T + bq) per head
  gemm_kernel<<<dim3(MROWS_X / 256, 8), 256, 0, stream>>>(
      xh, Wqh, bq, qh, nullptr, nullptr, 1);

  for (int i = 0; i < NI; ++i) {
    const _Float16* ei = eh + (size_t)i * MROWS_E * D_MODEL;
    // k = mask * softmax(emb @ Wk[i]^T + bk[i])
    gemm_kernel<<<dim3(MROWS_E / 256, 8), 256, 0, stream>>>(
        ei, Wkh + (size_t)i * D_MODEL * D_MODEL, bk + i * D_MODEL,
        kh + (size_t)i * MROWS_E * D_MODEL, nullptr, masks[i], 2);
    // v = emb @ Wv[i]^T + bv[i]
    gemm_kernel<<<dim3(MROWS_E / 256, 8), 256, 0, stream>>>(
        ei, Wvh + (size_t)i * D_MODEL * D_MODEL, bv + i * D_MODEL,
        vh + (size_t)i * MROWS_E * D_MODEL, nullptr, nullptr, 0);
  }

  kv_kernel<<<NI * BATCH * NHEAD, 256, 0, stream>>>(kh, vh, kvT);
  ksum_kernel<<<NI * BATCH * NHEAD, 256, 0, stream>>>(kh, ksum);
  dinv_kernel<<<(MROWS_X * NHEAD) / 256, 256, 0, stream>>>(qh, ksum, dinv);
  combine_kernel<<<(MROWS_X / 16) * NHEAD / 8, 256, 0, stream>>>(qh, kvT, dinv, oh);

  // final: out @ Wo^T + bo -> fp32
  gemm_kernel<<<dim3(MROWS_X / 256, 8), 256, 0, stream>>>(
      oh, Woh, bo, nullptr, (float*)d_out, nullptr, 3);
}
